// Decoder_5111011082645
// MI455X (gfx1250) — compile-verified, matrix operand order checked by reading, not run on hardware
//
#include <hip/hip_runtime.h>
#include <hip/hip_bf16.h>

// ---------------------------------------------------------------------------
// Sizes (fixed by the reference): B=4, N=256, H=128, edge_dim E=256, node D=384
// ---------------------------------------------------------------------------
#define B_   4
#define N_   256
#define H_   128
#define E_   256
#define D_   384
#define PITCH 257  // LDS row pitch (floats) -> bank (r+c)%64, conflict-free cols

typedef __attribute__((ext_vector_type(16))) __bf16 v16bf;
typedef __attribute__((ext_vector_type(8)))  float  v8f;

static __device__ __forceinline__ __bf16 f2bf(float x) { return (__bf16)x; }

// ---------------------------------------------------------------------------
// Kernel 0: init per-batch min keys (order-preserving uint encoding of f32)
// ---------------------------------------------------------------------------
__global__ void init_min_kernel(unsigned int* __restrict__ gmin) {
  if (threadIdx.x < B_) gmin[threadIdx.x] = 0xFFFFFFFFu;
}

// ---------------------------------------------------------------------------
// Kernel 1: p1 = node@W1+b1, p2 = node@W2+b2   (tiny: 0.2 GFLOP)
// grid (N, B), block 128 (one thread per h)
// ---------------------------------------------------------------------------
__global__ __launch_bounds__(H_) void node_proj_kernel(
    const float* __restrict__ node, const float* __restrict__ W1,
    const float* __restrict__ b1,   const float* __restrict__ W2,
    const float* __restrict__ b2,   float* __restrict__ p1,
    float* __restrict__ p2)
{
  __shared__ float row[D_];
  const int b = blockIdx.y, n = blockIdx.x, h = threadIdx.x;
  const float* src = node + ((size_t)(b * N_) + n) * D_;
  for (int d = h; d < D_; d += H_) row[d] = src[d];
  __syncthreads();
  float s1 = b1[h], s2 = b2[h];
  #pragma unroll 4
  for (int d = 0; d < D_; ++d) {
    const float x = row[d];
    s1 = fmaf(x, W1[d * H_ + h], s1);
    s2 = fmaf(x, W2[d * H_ + h], s2);
  }
  const size_t o = ((size_t)(b * N_) + n) * H_ + h;
  p1[o] = s1;
  p2[o] = s2;
}

// ---------------------------------------------------------------------------
// Kernel 2: fused edge GEMM (bf16 WMMA, f32 accum) + max + W4 dot + min
//   preds[b,i,j] = b4 + sum_h max(p1[b,i,h], p2[b,j,h]+b3[h]+p3[b,j,i,h])*W4[h]
// One workgroup per (b, j); 8 waves, wave w owns h-slice [16w, 16w+16).
// W3 bf16 fragments for all K preloaded once, reused across 16 i-tiles.
// ---------------------------------------------------------------------------
__global__ __launch_bounds__(256) void fused_edge_kernel(
    const float* __restrict__ edge, const float* __restrict__ p1,
    const float* __restrict__ p2,   const float* __restrict__ W3,
    const float* __restrict__ b3,   const float* __restrict__ W4,
    const float* __restrict__ b4,   float* __restrict__ preds,
    unsigned int* __restrict__ gmin)
{
  const int b    = blockIdx.y;
  const int j    = blockIdx.x;
  const int tid  = threadIdx.x;
  const int wave = tid >> 5;
  const int lane = tid & 31;
  const int hn   = lane & 15;          // N index within 16x16 tile
  const int h    = wave * 16 + hn;     // global h column this lane owns
  const int koff = (lane >> 4) * 8;    // K sub-offset per lane half (A & B layout)

  __shared__ float lds_sum[16];

  // --- Preload B fragments: W3 (E=256 x H=128) as bf16, 8 K-steps of 32 ---
  v16bf Bf[8];
  #pragma unroll
  for (int kk = 0; kk < 8; ++kk) {
    const int kb = kk * 32;
    #pragma unroll
    for (int p = 0; p < 8; ++p) {
      const int k0 = kb + ((p < 4) ? (koff + 2 * p) : (16 + koff + 2 * (p - 4)));
      Bf[kk][2 * p]     = f2bf(W3[(size_t)k0 * H_ + h]);
      Bf[kk][2 * p + 1] = f2bf(W3[(size_t)(k0 + 1) * H_ + h]);
    }
  }

  const float p2b3  = p2[((size_t)(b * N_) + j) * H_ + h] + b3[h];
  const float w4v   = W4[h];
  const float bias4 = b4[0];
  float lmin = 3.0e38f;

  const float* erow_base = edge + (((size_t)(b * N_) + j)) * (size_t)N_ * E_;

  for (int it = 0; it < 16; ++it) {
    const int i0 = it * 16;
    if (tid < 16) lds_sum[tid] = 0.0f;
    __syncthreads();

    // --- A fragment rows: edge_fts[b, j, i0 + (lane&15), :] ---
    const float* arow = erow_base + (size_t)(i0 + hn) * E_;

    v8f acc = {};
    #pragma unroll
    for (int kk = 0; kk < 8; ++kk) {
      const int kb = kk * 32 + koff;
      const float4 a0 = *(const float4*)(arow + kb);
      const float4 a1 = *(const float4*)(arow + kb + 4);
      const float4 a2 = *(const float4*)(arow + kb + 16);
      const float4 a3 = *(const float4*)(arow + kb + 20);
      v16bf A;
      A[0]  = f2bf(a0.x); A[1]  = f2bf(a0.y); A[2]  = f2bf(a0.z); A[3]  = f2bf(a0.w);
      A[4]  = f2bf(a1.x); A[5]  = f2bf(a1.y); A[6]  = f2bf(a1.z); A[7]  = f2bf(a1.w);
      A[8]  = f2bf(a2.x); A[9]  = f2bf(a2.y); A[10] = f2bf(a2.z); A[11] = f2bf(a2.w);
      A[12] = f2bf(a3.x); A[13] = f2bf(a3.y); A[14] = f2bf(a3.z); A[15] = f2bf(a3.w);
      acc = __builtin_amdgcn_wmma_f32_16x16x32_bf16(
          /*neg_a=*/false, A, /*neg_b=*/false, Bf[kk],
          /*c_mod=*/(short)0, acc, /*reuse_a=*/false, /*reuse_b=*/false);
    }

    // --- combine: C vgpr r -> row i0+r (lanes 0-15) / i0+r+8 (lanes 16-31) ---
    const int mbase = i0 + ((lane >> 4) << 3);
    #pragma unroll
    for (int r = 0; r < 8; ++r) {
      const int irow = mbase + r;
      const float p1v = p1[((size_t)(b * N_) + irow) * H_ + h];
      float v = fmaxf(p1v, p2b3 + acc[r]) * w4v;
      // reduce the 16 h-lanes of this half-wave
      v += __shfl_xor(v, 1, 32);
      v += __shfl_xor(v, 2, 32);
      v += __shfl_xor(v, 4, 32);
      v += __shfl_xor(v, 8, 32);
      if (hn == 0) atomicAdd(&lds_sum[(lane >> 4) * 8 + r], v);  // ds_add_f32
    }
    __syncthreads();

    if (tid < 16) {
      const float pv = lds_sum[tid] + bias4;
      preds[((size_t)(b * N_) + (i0 + tid)) * N_ + j] = pv;  // transposed store
      lmin = fminf(lmin, pv);
    }
  }

  // --- per-batch global min (order-preserving uint atomicMin) ---
  if (tid < 16) {
    float v = lmin;
    v = fminf(v, __shfl_xor(v, 1, 32));
    v = fminf(v, __shfl_xor(v, 2, 32));
    v = fminf(v, __shfl_xor(v, 4, 32));
    v = fminf(v, __shfl_xor(v, 8, 32));
    if (tid == 0) {
      const unsigned int u = __float_as_uint(v);
      const unsigned int key = (u & 0x80000000u) ? ~u : (u | 0x80000000u);
      atomicMin(&gmin[b], key);
    }
  }
}

// ---------------------------------------------------------------------------
// Kernel 3: mask + temperature + diag, then 10x (row LSE, col LSE), all in LDS.
// One 1024-thread WG per batch; 256x257-pitched f32 matrix = 263,168 B of LDS
// (CDNA5: up to 320 KB per workgroup). Pitch 257 => column pass conflict-free.
// ---------------------------------------------------------------------------
__global__ __launch_bounds__(1024) void sinkhorn_kernel(
    const float* __restrict__ preds, const float* __restrict__ adj,
    const unsigned int* __restrict__ gmin, float* __restrict__ out)
{
  extern __shared__ float sm[];
  const int b    = blockIdx.x;
  const int tid  = threadIdx.x;
  const int wave = tid >> 5;
  const int lane = tid & 31;

  const unsigned int key = gmin[b];
  const unsigned int u   = (key & 0x80000000u) ? (key & 0x7FFFFFFFu) : ~key;
  const float pmin = __uint_as_float(u);
  const float fill = fminf(-1.0f, pmin - 1.0f);

  const size_t base = (size_t)b * N_ * N_;
  for (int e = tid; e < N_ * N_; e += 1024) {
    const int i = e >> 8, jc = e & 255;
    float v = preds[base + e];
    const float a = adj[base + e];
    v = (a > 0.5f) ? v : fill;
    v *= 10.0f;                       // x / temperature, temp = 0.1
    if (i == jc) v = -1.0e6f;         // diagonal mask (applied after /temp)
    sm[i * PITCH + jc] = v;
  }
  __syncthreads();

  for (int iter = 0; iter < 10; ++iter) {
    // ---- log_softmax over rows (axis=-1): wave w handles rows 8w..8w+7 ----
    #pragma unroll 1
    for (int rr = 0; rr < 8; ++rr) {
      const int r = wave * 8 + rr;
      float vv[8];
      float m = -3.0e38f;
      #pragma unroll
      for (int t = 0; t < 8; ++t) {
        vv[t] = sm[r * PITCH + lane + t * 32];
        m = fmaxf(m, vv[t]);
      }
      m = fmaxf(m, __shfl_xor(m, 1, 32));
      m = fmaxf(m, __shfl_xor(m, 2, 32));
      m = fmaxf(m, __shfl_xor(m, 4, 32));
      m = fmaxf(m, __shfl_xor(m, 8, 32));
      m = fmaxf(m, __shfl_xor(m, 16, 32));
      float s = 0.0f;
      #pragma unroll
      for (int t = 0; t < 8; ++t) s += __expf(vv[t] - m);
      s += __shfl_xor(s, 1, 32);
      s += __shfl_xor(s, 2, 32);
      s += __shfl_xor(s, 4, 32);
      s += __shfl_xor(s, 8, 32);
      s += __shfl_xor(s, 16, 32);
      const float lse = m + __logf(s);
      #pragma unroll
      for (int t = 0; t < 8; ++t) sm[r * PITCH + lane + t * 32] = vv[t] - lse;
    }
    __syncthreads();

    // ---- log_softmax over cols (axis=-2): wave w handles cols 8w..8w+7 ----
    #pragma unroll 1
    for (int cc = 0; cc < 8; ++cc) {
      const int c = wave * 8 + cc;
      float vv[8];
      float m = -3.0e38f;
      #pragma unroll
      for (int t = 0; t < 8; ++t) {
        vv[t] = sm[(lane + t * 32) * PITCH + c];
        m = fmaxf(m, vv[t]);
      }
      m = fmaxf(m, __shfl_xor(m, 1, 32));
      m = fmaxf(m, __shfl_xor(m, 2, 32));
      m = fmaxf(m, __shfl_xor(m, 4, 32));
      m = fmaxf(m, __shfl_xor(m, 8, 32));
      m = fmaxf(m, __shfl_xor(m, 16, 32));
      float s = 0.0f;
      #pragma unroll
      for (int t = 0; t < 8; ++t) s += __expf(vv[t] - m);
      s += __shfl_xor(s, 1, 32);
      s += __shfl_xor(s, 2, 32);
      s += __shfl_xor(s, 4, 32);
      s += __shfl_xor(s, 8, 32);
      s += __shfl_xor(s, 16, 32);
      const float lse = m + __logf(s);
      #pragma unroll
      for (int t = 0; t < 8; ++t) sm[(lane + t * 32) * PITCH + c] = vv[t] - lse;
    }
    __syncthreads();
  }

  for (int e = tid; e < N_ * N_; e += 1024) {
    const int i = e >> 8, jc = e & 255;
    out[base + e] = sm[i * PITCH + jc];
  }
}

// ---------------------------------------------------------------------------
// Host launcher
// ---------------------------------------------------------------------------
extern "C" void kernel_launch(void* const* d_in, const int* in_sizes, int n_in,
                              void* d_out, int out_size, void* d_ws, size_t ws_size,
                              hipStream_t stream) {
  const float* node = (const float*)d_in[0];
  const float* edge = (const float*)d_in[1];
  const float* adj  = (const float*)d_in[2];
  const float* W1   = (const float*)d_in[3];
  const float* b1   = (const float*)d_in[4];
  const float* W2   = (const float*)d_in[5];
  const float* b2   = (const float*)d_in[6];
  const float* W3   = (const float*)d_in[7];
  const float* b3   = (const float*)d_in[8];
  const float* W4   = (const float*)d_in[9];
  const float* b4   = (const float*)d_in[10];

  // workspace layout (floats): p1 | p2 | preds | gmin(uint)
  float* p1    = (float*)d_ws;
  float* p2    = p1 + (size_t)B_ * N_ * H_;
  float* preds = p2 + (size_t)B_ * N_ * H_;
  unsigned int* gmin = (unsigned int*)(preds + (size_t)B_ * N_ * N_);

  hipLaunchKernelGGL(init_min_kernel, dim3(1), dim3(32), 0, stream, gmin);

  hipLaunchKernelGGL(node_proj_kernel, dim3(N_, B_), dim3(H_), 0, stream,
                     node, W1, b1, W2, b2, p1, p2);

  hipLaunchKernelGGL(fused_edge_kernel, dim3(N_, B_), dim3(256), 0, stream,
                     edge, p1, p2, W3, b3, W4, b4, preds, gmin);

  hipLaunchKernelGGL(sinkhorn_kernel, dim3(B_), dim3(1024),
                     (size_t)(N_ * PITCH * sizeof(float)), stream,
                     preds, adj, gmin, (float*)d_out);
}
